// DS_Block_28028956574157
// MI455X (gfx1250) — matrix operand order repaired and checked by reading, not compiled, not run on hardware
//
#include <hip/hip_runtime.h>
#include <hip/hip_bf16.h>
#include <math.h>

// ---------------------------------------------------------------------------
// CDNA5 (gfx1250) implementation. wave32. Matrix ops via v_wmma_f32_16x16x32_f16.
// ---------------------------------------------------------------------------

typedef __attribute__((ext_vector_type(16))) _Float16 v16h;
typedef __attribute__((ext_vector_type(2)))  _Float16 v2h;
typedef __attribute__((ext_vector_type(8)))  float    v8f;

#define EPSV 1e-5f

// ===========================================================================
// WMMA GEMM:  OUT[M,N] = X[M,K] * W[N,K]^T  (+bias) (+BN) (+add) (+relu/gelu)
// flags: 1=relu, 2=gelu(exact), 4=bn affine (g,b,m,v per output column), 8=add
// Block = 128 threads = 4 waves; block tile 64x64; wave tile 16x64; K step 32.
//
// - LDS holds FRAGMENT-READY f16 (Af[mtile][lane][16]) so a fragment load is
//   one 32-byte read (2x ds_load_b128).
// - Double-buffered LDS + software pipeline: stage tile t+1 while WMMAs
//   consume tile t. One barrier per K-step.
// - GUARD template: interior blocks take an unguarded staging path (batched
//   float2 loads, no exec-mask per element, no per-load waits); only edge
//   blocks / K-tail pay for bounds checks.
// - f32->f16 conversion must go through VALU (TDM/async movers copy raw
//   bytes), so cvt-through-LDS staging is the correct data path.
//
// Fragment layouts per CDNA5 ISA 7.12.2 (wave32):
//   A 16x32 f16 : lane = (m&15) + 16*((k>>3)&1); e = (k&7) + (k>=16 ? 8 : 0)
//   B 32x16 f16 : lane = (n&15) + 16*((k>>3)&1); e = (k&7) + (k>=16 ? 8 : 0)
//   C/D 16x16 f32: lane<16 -> N=lane, r->M=r ; lane>=16 -> N=lane-16, r->M=8+r
// ===========================================================================
template <bool GUARD>
__device__ __forceinline__ void stage_tile(
    const float* __restrict__ X, const float* __restrict__ W,
    _Float16* __restrict__ Af, _Float16* __restrict__ Bf,
    int bm, int bn0, int M, int N, int K, int k0, int tid)
{
  // ---- A tile: 64 rows x 16 k-pairs ----
#pragma unroll
  for (int it = 0; it < 8; it++) {
    int i = tid + it * 128;
    int m = i >> 4;
    int k = (i & 15) * 2;
    float2 xv;
    if (GUARD) {
      int gr = bm + m, gc = k0 + k;
      xv = make_float2(0.f, 0.f);
      if (gr < M && gc < K) xv = *(const float2*)&X[(long)gr * K + gc];
    } else {
      xv = *(const float2*)&X[(long)(bm + m) * K + (k0 + k)];
    }
    int laneA = (m & 15) + (((k >> 3) & 1) << 4);
    int eA    = (k & 7) + ((k & 16) ? 8 : 0);
    v2h p; p[0] = (_Float16)xv.x; p[1] = (_Float16)xv.y;
    *(v2h*)&Af[((m >> 4) * 32 + laneA) * 16 + eA] = p;
  }
  // ---- B tile: W[bn0+n][k0+k] ----
#pragma unroll
  for (int it = 0; it < 8; it++) {
    int i = tid + it * 128;
    int n = i >> 4;
    int k = (i & 15) * 2;
    float2 wv;
    if (GUARD) {
      int gn = bn0 + n, gc = k0 + k;
      wv = make_float2(0.f, 0.f);
      if (gn < N && gc < K) wv = *(const float2*)&W[(long)gn * K + gc];
    } else {
      wv = *(const float2*)&W[(long)(bn0 + n) * K + (k0 + k)];
    }
    int laneB = (n & 15) + (((k >> 3) & 1) << 4);
    int eB    = (k & 7) + ((k & 16) ? 8 : 0);
    v2h p; p[0] = (_Float16)wv.x; p[1] = (_Float16)wv.y;
    *(v2h*)&Bf[((n >> 4) * 32 + laneB) * 16 + eB] = p;
  }
}

__global__ __launch_bounds__(128)
void k_gemm(const float* __restrict__ X, const float* __restrict__ W,
            const float* __restrict__ bias, const float* __restrict__ addsrc,
            const float* __restrict__ bng, const float* __restrict__ bnb,
            const float* __restrict__ bnm, const float* __restrict__ bnv,
            float* __restrict__ OUT, int M, int N, int K, int flags)
{
  __shared__ __align__(32) _Float16 Af[2][4 * 32 * 16];  // double-buffered
  __shared__ __align__(32) _Float16 Bf[2][4 * 32 * 16];

  const int tid  = threadIdx.x;
  const int lane = tid & 31;
  const int wave = tid >> 5;
  const int bm   = blockIdx.y * 64;
  const int bn0  = blockIdx.x * 64;

  const bool interior = (bm + 64 <= M) && (bn0 + 64 <= N);
  const int  nk = (K + 31) >> 5;

  v8f acc[4] = {};

  // prologue: stage tile 0 into buffer 0
  if (interior && 32 <= K)
    stage_tile<false>(X, W, Af[0], Bf[0], bm, bn0, M, N, K, 0, tid);
  else
    stage_tile<true>(X, W, Af[0], Bf[0], bm, bn0, M, N, K, 0, tid);
  __syncthreads();

  for (int t = 0; t < nk; t++) {
    const int cur = t & 1;
    const int k0n = (t + 1) * 32;
    // stage next tile into alternate buffer while computing this one
    if (t + 1 < nk) {
      if (interior && k0n + 32 <= K)
        stage_tile<false>(X, W, Af[cur ^ 1], Bf[cur ^ 1], bm, bn0, M, N, K, k0n, tid);
      else
        stage_tile<true>(X, W, Af[cur ^ 1], Bf[cur ^ 1], bm, bn0, M, N, K, k0n, tid);
    }
    // prefetch tile t+2 (global_prefetch_b8)
    if ((t + 2) * 32 < K) {
      int gr = bm + (tid >> 1);
      if (gr < M) __builtin_prefetch(&X[(long)gr * K + (t + 2) * 32], 0, 1);
    }

    // ---- 4 WMMAs per wave: one A fragment vs 4 B fragments ----
    v16h a = *(const v16h*)&Af[cur][(wave * 32 + lane) * 16];
#pragma unroll
    for (int nt = 0; nt < 4; nt++) {
      v16h b = *(const v16h*)&Bf[cur][(nt * 32 + lane) * 16];
      acc[nt] = __builtin_amdgcn_wmma_f32_16x16x32_f16(
          false, a, false, b, (short)0, acc[nt], false, false);
    }
    __syncthreads();
  }

  // ---- epilogue ----
  const int ncol  = lane & 15;
  const int mbase = (lane >> 4) * 8;
#pragma unroll
  for (int nt = 0; nt < 4; nt++) {
#pragma unroll
    for (int r = 0; r < 8; r++) {
      int gm = bm + wave * 16 + mbase + r;
      int gn = bn0 + nt * 16 + ncol;
      if (gm < M && gn < N) {
        float v = acc[nt][r];
        if (bias) v += bias[gn];
        if (flags & 4) v = (v - bnm[gn]) * (bng[gn] * rsqrtf(bnv[gn] + EPSV)) + bnb[gn];
        if (flags & 8) v += addsrc[(long)gm * N + gn];
        if (flags & 1) v = v > 0.f ? v : 0.f;
        if (flags & 2) v = 0.5f * v * (1.f + erff(v * 0.70710678118654752f));
        OUT[(long)gm * N + gn] = v;
      }
    }
  }
}

// ===========================================================================
// conv0: out[(b*N+n), o] = relu(bn( sum_{i<4} x[(b*N+n)*4+i]*w[o*4+i] + b0[o] ))
// ===========================================================================
__global__ void k_conv0(const float* __restrict__ x, const float* __restrict__ w,
                        const float* __restrict__ b0,
                        const float* __restrict__ g, const float* __restrict__ bb,
                        const float* __restrict__ m, const float* __restrict__ v,
                        float* __restrict__ out, long total)
{
  long e = (long)blockIdx.x * blockDim.x + threadIdx.x;
  if (e >= total) return;
  long row = e >> 7;           // /128
  int  o   = (int)(e & 127);
  float s = b0[o];
#pragma unroll
  for (int i = 0; i < 4; i++) s += x[row * 4 + i] * w[o * 4 + i];
  s = (s - m[o]) * (g[o] * rsqrtf(v[o] + EPSV)) + bb[o];
  out[e] = s > 0.f ? s : 0.f;
}

// ===========================================================================
// InstanceNorm (over n per (b,c)) + BN affine (+residual add) (+relu)
// flags: 1=relu at end, 2=add residual before relu
// grid = B*C blocks, 256 threads
// ===========================================================================
__global__ __launch_bounds__(256)
void k_inorm_bn(const float* __restrict__ X,
                const float* __restrict__ g, const float* __restrict__ bb,
                const float* __restrict__ m, const float* __restrict__ v,
                const float* __restrict__ res, float* __restrict__ OUT,
                int Nn, int Cn, int flags)
{
  int bc = blockIdx.x;
  int b = bc / Cn, c = bc % Cn;
  __shared__ float s1[256], s2[256];
  float sum = 0.f, sq = 0.f;
  for (int n = threadIdx.x; n < Nn; n += 256) {
    float x = X[((long)b * Nn + n) * Cn + c];
    sum += x; sq += x * x;
  }
  s1[threadIdx.x] = sum; s2[threadIdx.x] = sq; __syncthreads();
  for (int s = 128; s > 0; s >>= 1) {
    if (threadIdx.x < s) { s1[threadIdx.x] += s1[threadIdx.x + s]; s2[threadIdx.x] += s2[threadIdx.x + s]; }
    __syncthreads();
  }
  float mean = s1[0] / Nn;
  float var  = s2[0] / Nn - mean * mean;
  float rs   = rsqrtf(var + EPSV);
  float sc   = g[c] * rsqrtf(v[c] + EPSV);
  for (int n = threadIdx.x; n < Nn; n += 256) {
    long off = ((long)b * Nn + n) * Cn + c;
    float x = (X[off] - mean) * rs;
    x = (x - m[c]) * sc + bb[c];
    if (flags & 2) x += res[off];
    if (flags & 1) x = x > 0.f ? x : 0.f;
    OUT[off] = x;
  }
}

// ===========================================================================
// LayerNorm over C=128 per row. grid = rows, block = 128
// ===========================================================================
__global__ __launch_bounds__(128)
void k_layernorm(const float* __restrict__ X, const float* __restrict__ g,
                 const float* __restrict__ b, float* __restrict__ OUT)
{
  long row = blockIdx.x;
  int c = threadIdx.x;
  __shared__ float r1[128], r2[128];
  float x = X[row * 128 + c];
  r1[c] = x; r2[c] = x * x; __syncthreads();
  for (int s = 64; s > 0; s >>= 1) {
    if (c < s) { r1[c] += r1[c + s]; r2[c] += r2[c + s]; }
    __syncthreads();
  }
  float m = r1[0] * (1.f / 128.f);
  float v = r2[0] * (1.f / 128.f) - m * m;
  OUT[row * 128 + c] = (x - m) * rsqrtf(v + EPSV) * g[c] + b[c];
}

// ===========================================================================
// assemble token buffer: t[b*T+0] = cls; t[b*T+1+n] = pe[(b*N+n)]
// ===========================================================================
__global__ void k_assemble_t(const float* __restrict__ pe, const float* __restrict__ cls,
                             float* __restrict__ t, int Nn, int Tt, long total)
{
  long e = (long)blockIdx.x * blockDim.x + threadIdx.x;
  if (e >= total) return;
  long row = e >> 7;
  int  c   = (int)(e & 127);
  int  b   = (int)(row / Tt);
  int  p   = (int)(row % Tt);
  t[e] = (p == 0) ? cls[c] : pe[((long)b * Nn + (p - 1)) * 128 + c];
}

__global__ void k_strip_cls(const float* __restrict__ t, float* __restrict__ Z,
                            int Nn, int Tt, long total)
{
  long e = (long)blockIdx.x * blockDim.x + threadIdx.x;
  if (e >= total) return;
  long row = e >> 7;
  int  c   = (int)(e & 127);
  int  b   = (int)(row / Nn);
  int  n   = (int)(row % Nn);
  Z[e] = t[((long)b * Tt + 1 + n) * 128 + c];
}

// ===========================================================================
// Attention: one block per (b,h,i). qkv [B*T,384] (q|k|v, each h*64+d).
// Streaming softmax with scores cached in LDS.
// ===========================================================================
__global__ __launch_bounds__(256)
void k_attention(const float* __restrict__ qkv, float* __restrict__ o, int Tt)
{
  const int gi = blockIdx.x;
  const int i  = gi % Tt;
  const int bh = gi / Tt;
  const int h  = bh & 1;
  const int b  = bh >> 1;
  const int tid = threadIdx.x;
  __shared__ float q[64];
  __shared__ float sc[2016];
  __shared__ float red[256];
  __shared__ float op[4][64];
  const long base = (long)b * Tt * 384;
  if (tid < 64) q[tid] = qkv[base + (long)i * 384 + h * 64 + tid];
  __syncthreads();
  float lmax = -3.4e38f;
  for (int j = tid; j < Tt; j += 256) {
    const float* kr = &qkv[base + (long)j * 384 + 128 + h * 64];
    float s = 0.f;
#pragma unroll 8
    for (int d = 0; d < 64; d++) s += q[d] * kr[d];
    s *= 0.125f;          // DIM_HEAD^-0.5
    sc[j] = s;
    lmax = fmaxf(lmax, s);
  }
  red[tid] = lmax; __syncthreads();
  for (int s = 128; s > 0; s >>= 1) { if (tid < s) red[tid] = fmaxf(red[tid], red[tid + s]); __syncthreads(); }
  float mx = red[0]; __syncthreads();
  float lsum = 0.f;
  for (int j = tid; j < Tt; j += 256) {
    float p = expf(sc[j] - mx);
    sc[j] = p; lsum += p;
  }
  red[tid] = lsum; __syncthreads();
  for (int s = 128; s > 0; s >>= 1) { if (tid < s) red[tid] += red[tid + s]; __syncthreads(); }
  float inv = 1.f / red[0];
  const int d = tid & 63;
  const int part = tid >> 6;
  float a = 0.f;
  for (int j = part; j < Tt; j += 4)
    a += sc[j] * qkv[base + (long)j * 384 + 256 + h * 64 + d];
  op[part][d] = a; __syncthreads();
  if (tid < 64) {
    float s = (op[0][tid] + op[1][tid] + op[2][tid] + op[3][tid]) * inv;
    o[((long)b * Tt + i) * 128 + h * 64 + tid] = s;
  }
}

// ===========================================================================
// DGCNN: top-K=9 neighbors by pd = 2 x_i.x_j - |x_i|^2 - |x_j|^2 (per b row i)
// ===========================================================================
__global__ __launch_bounds__(128)
void k_topk(const float* __restrict__ X, int* __restrict__ nbr, int Nn)
{
  const int gi = blockIdx.x;
  const int b = gi / Nn, i = gi % Nn;
  const int tid = threadIdx.x;
  __shared__ float xi[128];
  xi[tid] = X[((long)b * Nn + i) * 128 + tid];
  __syncthreads();
  float xi2 = 0.f;
  for (int c = 0; c < 128; c++) xi2 += xi[c] * xi[c];
  float bv[9]; int bi[9];
#pragma unroll
  for (int k = 0; k < 9; k++) { bv[k] = -3.4e38f; bi[k] = 0x7fffffff; }
  for (int j = tid; j < Nn; j += 128) {
    const float* xj = &X[((long)b * Nn + j) * 128];
    float dot = 0.f, xx = 0.f;
    for (int c = 0; c < 128; c++) { float t = xj[c]; dot += xi[c] * t; xx += t * t; }
    float pd = 2.f * dot - xi2 - xx;
    if (pd > bv[8] || (pd == bv[8] && j < bi[8])) {
      bv[8] = pd; bi[8] = j;
#pragma unroll
      for (int k = 8; k > 0; k--) {
        if (bv[k] > bv[k - 1] || (bv[k] == bv[k - 1] && bi[k] < bi[k - 1])) {
          float tv = bv[k]; bv[k] = bv[k - 1]; bv[k - 1] = tv;
          int ti = bi[k]; bi[k] = bi[k - 1]; bi[k - 1] = ti;
        }
      }
    }
  }
  __shared__ float cv[128 * 9];
  __shared__ int   ci[128 * 9];
#pragma unroll
  for (int k = 0; k < 9; k++) { cv[tid * 9 + k] = bv[k]; ci[tid * 9 + k] = bi[k]; }
  __syncthreads();
  if (tid == 0) {
    for (int k = 0; k < 9; k++) {
      int best = 0;
      for (int q = 1; q < 128 * 9; q++)
        if (cv[q] > cv[best] || (cv[q] == cv[best] && ci[q] < ci[best])) best = q;
      nbr[((long)b * Nn + i) * 9 + k] = ci[best];
      cv[best] = -3.4e38f;
    }
  }
}

// Edge features: G2[((b*N+n)*3+j), c*3+t]; c<128 -> ctr; c>=128 -> ctr-nbr
__global__ void k_build_edge(const float* __restrict__ X, const int* __restrict__ nbr,
                             float* __restrict__ G2, int Nn, long total)
{
  long e = (long)blockIdx.x * blockDim.x + threadIdx.x;
  if (e >= total) return;
  long r = e / 768;
  int  q = (int)(e % 768);
  int  c = q / 3, t = q % 3;
  int  j = (int)(r % 3);
  long row = r / 3;
  int  b = (int)(row / Nn);
  float val;
  if (c < 128) {
    val = X[row * 128 + c];
  } else {
    int cc = c - 128;
    int nb = nbr[row * 9 + (j * 3 + t)];
    val = X[row * 128 + cc] - X[((long)b * Nn + nb) * 128 + cc];
  }
  G2[e] = val;
}

// H2[row, c*3+t] = H1[row*3+t, c]
__global__ void k_pack_h2(const float* __restrict__ H1, float* __restrict__ H2, long total)
{
  long e = (long)blockIdx.x * blockDim.x + threadIdx.x;
  if (e >= total) return;
  long row = e / 384;
  int  q = (int)(e % 384);
  int  c = q / 3, t = q % 3;
  H2[e] = H1[(row * 3 + t) * 128 + c];
}

// ===========================================================================
// GEMV (lin0/lin1): out[row] = dot(X[row,:C], w) + b
// ===========================================================================
__global__ void k_gemv(const float* __restrict__ X, const float* __restrict__ w,
                       const float* __restrict__ b, float* __restrict__ out, int M)
{
  int r = blockIdx.x * blockDim.x + threadIdx.x;
  if (r >= M) return;
  float s = 0.f;
  for (int k = 0; k < 128; k++) s += X[(long)r * 128 + k] * w[k];
  out[r] = s + b[0];
}

// ===========================================================================
// GCN closed form: L = rr_i rr_j wa_i wa_j + rr_i^2 delta_ij,
// rr_i = rsqrt(wa_i * S + 1), S = sum_j wa_j, wa = relu(tanh(w_l))
// ===========================================================================
__global__ __launch_bounds__(256)
void k_gcn_prep(const float* __restrict__ wl, float* __restrict__ wa,
                float* __restrict__ rr, int Nn)
{
  int b = blockIdx.x;
  __shared__ float red[256];
  float s = 0.f;
  for (int j = threadIdx.x; j < Nn; j += 256) {
    float w = tanhf(wl[(long)b * Nn + j]);
    w = w > 0.f ? w : 0.f;
    wa[(long)b * Nn + j] = w;
    s += w;
  }
  red[threadIdx.x] = s; __syncthreads();
  for (int t = 128; t > 0; t >>= 1) { if (threadIdx.x < t) red[threadIdx.x] += red[threadIdx.x + t]; __syncthreads(); }
  float S = red[0];
  for (int j = threadIdx.x; j < Nn; j += 256)
    rr[(long)b * Nn + j] = rsqrtf(wa[(long)b * Nn + j] * S + 1.f);
}

__global__ __launch_bounds__(256)
void k_gcn_tc(const float* __restrict__ X, const float* __restrict__ wa,
              const float* __restrict__ rr, float* __restrict__ tc, int Nn)
{
  int blk = blockIdx.x;
  int b = blk >> 7, c = blk & 127;
  __shared__ float red[256];
  float s = 0.f;
  for (int j = threadIdx.x; j < Nn; j += 256) {
    long row = (long)b * Nn + j;
    s += wa[row] * rr[row] * X[row * 128 + c];
  }
  red[threadIdx.x] = s; __syncthreads();
  for (int t = 128; t > 0; t >>= 1) { if (threadIdx.x < t) red[threadIdx.x] += red[threadIdx.x + t]; __syncthreads(); }
  if (threadIdx.x == 0) tc[blk] = red[0];
}

__global__ void k_gcn_apply(const float* __restrict__ X, const float* __restrict__ wa,
                            const float* __restrict__ rr, const float* __restrict__ tc,
                            float* __restrict__ OUT, int Nn, long total)
{
  long e = (long)blockIdx.x * blockDim.x + threadIdx.x;
  if (e >= total) return;
  long row = e >> 7;
  int  c   = (int)(e & 127);
  int  b   = (int)(row / Nn);
  float r = rr[row], w = wa[row];
  OUT[e] = w * r * tc[b * 128 + c] + r * r * X[e];
}

// ===========================================================================
// misc elementwise
// ===========================================================================
__global__ void k_add(const float* __restrict__ a, const float* __restrict__ b,
                      float* __restrict__ o, long total)
{
  long e = (long)blockIdx.x * blockDim.x + threadIdx.x;
  if (e >= total) return;
  o[e] = a[e] + b[e];
}

__global__ void k_concat2(const float* __restrict__ A, const float* __restrict__ Bp,
                          float* __restrict__ C, long total)
{
  long e = (long)blockIdx.x * blockDim.x + threadIdx.x;
  if (e >= total) return;
  long row = e >> 8;            // /256
  int  c   = (int)(e & 255);
  C[e] = (c < 128) ? A[row * 128 + c] : Bp[row * 128 + (c - 128)];
}

// ===========================================================================
// Final: stable descending rank selection (matches argsort(-w) + take nh)
// ===========================================================================
__global__ __launch_bounds__(256)
void k_select(const float* __restrict__ wg, const float* __restrict__ y,
              const float* __restrict__ x4, float* __restrict__ xds,
              float* __restrict__ yds, float* __restrict__ wds, int Nn, int nh)
{
  int b = blockIdx.x;
  for (int i = threadIdx.x; i < Nn; i += blockDim.x) {
    float wi = wg[(long)b * Nn + i];
    int rank = 0;
    for (int j = 0; j < Nn; j++) {
      float wj = wg[(long)b * Nn + j];
      if (wj > wi || (wj == wi && j < i)) rank++;
    }
    if (rank < nh) {
      wds[(long)b * nh + rank] = wi;
      yds[(long)b * nh + rank] = y[(long)b * Nn + i];
#pragma unroll
      for (int q = 0; q < 4; q++)
        xds[((long)b * nh + rank) * 4 + q] = x4[((long)b * Nn + i) * 4 + q];
    }
  }
}

// ===========================================================================
// Host orchestration
// ===========================================================================
extern "C" void kernel_launch(void* const* d_in, const int* in_sizes, int n_in,
                              void* d_out, int out_size, void* d_ws, size_t ws_size,
                              hipStream_t stream)
{
  (void)in_sizes; (void)n_in; (void)out_size; (void)ws_size;
  const int Bn = 8, Nn = 2000, Tt = 2001;
  const long NT = (long)Bn * Nn;          // 16000
  const long MT = (long)Bn * Tt;          // 16008

  auto P = [&](int i) { return (const float*)d_in[i]; };

  // ---- workspace bump allocator (floats) ----
  float* wsf = (float*)d_ws;
  long off = 0;
  auto alloc = [&](long nf) { float* p = wsf + off; off += nf; return p; };

  float* cf    = alloc(NT * 128);
  float* cfB   = alloc(NT * 128);
  float* rt1   = alloc(NT * 128);
  float* rt2   = alloc(NT * 128);
  float* zb    = alloc(NT * 128);
  float* ol    = alloc(NT * 128);
  float* olB   = alloc(NT * 128);
  float* catb  = alloc(NT * 256);
  float* l1    = alloc(NT * 128);
  float* tm2b  = alloc(NT * 128);
  float* goutb = alloc(NT * 128);
  float* fgb   = alloc(NT * 128);
  float* fmb   = alloc(NT * 128);
  float* ogb   = alloc(NT * 128);
  float* g1b   = alloc(NT * 128);
  float* tm3b  = alloc(NT * 128);
  float* g2b   = alloc(NT * 128);
  float* petmp = alloc(NT * 128);
  float* tbuf  = alloc(MT * 128);
  float* tln   = alloc(MT * 128);
  float* attno = alloc(MT * 128);
  float* qkvb  = alloc(MT * 384);
  float* mlpb  = alloc(MT * 2084);
  float* wlb   = alloc(NT);
  float* wab   = alloc(NT);
  float* rrb   = alloc(NT);
  float* tcb   = alloc(Bn * 128);
  int*   nbr   = (int*)alloc(NT * 9);
  float* G2b   = alloc(NT * 3 * 768);
  float* H1b   = alloc(NT * 3 * 128);
  float* H2b   = alloc(NT * 384);

  auto gemm = [&](const float* X, const float* W, const float* bias, const float* add,
                  const float* g, const float* bb, const float* m, const float* v,
                  float* OUT, int M, int N, int K, int fl) {
    dim3 gr((N + 63) / 64, (M + 63) / 64);
    k_gemm<<<gr, 128, 0, stream>>>(X, W, bias, add, g, bb, m, v, OUT, M, N, K, fl);
  };
  auto ew = [&](long total) { return dim3((unsigned)((total + 255) / 256)); };

  // resblock: params at base p = {w1,b1, bn1(g,b,m,v), w2,b2, bn2(g,b,m,v)}
  auto resblock = [&](const float* in, float* out, int p) {
    gemm(in, P(p), P(p + 1), nullptr, nullptr, nullptr, nullptr, nullptr,
         rt1, (int)NT, 128, 128, 0);
    k_inorm_bn<<<Bn * 128, 256, 0, stream>>>(rt1, P(p + 2), P(p + 3), P(p + 4), P(p + 5),
                                             nullptr, rt2, Nn, 128, 1);
    gemm(rt2, P(p + 6), P(p + 7), nullptr, nullptr, nullptr, nullptr, nullptr,
         rt1, (int)NT, 128, 128, 0);
    k_inorm_bn<<<Bn * 128, 256, 0, stream>>>(rt1, P(p + 8), P(p + 9), P(p + 10), P(p + 11),
                                             in, out, Nn, 128, 3);
  };

  // transformer module: tm params pe_w=56, pe_b=57, cls=58, layers at 59+l*11
  auto transm = [&](const float* in, float* outZ) {
    gemm(in, P(56), P(57), nullptr, nullptr, nullptr, nullptr, nullptr,
         petmp, (int)NT, 128, 128, 0);
    k_assemble_t<<<ew(MT * 128), 256, 0, stream>>>(petmp, P(58), tbuf, Nn, Tt, MT * 128);
    for (int l = 0; l < 2; l++) {
      int bp = 59 + l * 11;  // ln1g,ln1b,wqkv,wo,bo,ln2g,ln2b,wf1,bf1,wf2,bf2
      k_layernorm<<<(unsigned)MT, 128, 0, stream>>>(tbuf, P(bp), P(bp + 1), tln);
      gemm(tln, P(bp + 2), nullptr, nullptr, nullptr, nullptr, nullptr, nullptr,
           qkvb, (int)MT, 384, 128, 0);
      k_attention<<<Bn * 2 * Tt, 256, 0, stream>>>(qkvb, attno, Tt);
      gemm(attno, P(bp + 3), P(bp + 4), tbuf, nullptr, nullptr, nullptr, nullptr,
           tbuf, (int)MT, 128, 128, 8);
      k_layernorm<<<(unsigned)MT, 128, 0, stream>>>(tbuf, P(bp + 5), P(bp + 6), tln);
      gemm(tln, P(bp + 7), P(bp + 8), nullptr, nullptr, nullptr, nullptr, nullptr,
           mlpb, (int)MT, 2084, 128, 2);
      gemm(mlpb, P(bp + 9), P(bp + 10), tbuf, nullptr, nullptr, nullptr, nullptr,
           tbuf, (int)MT, 128, 2084, 8);
    }
    k_strip_cls<<<ew(NT * 128), 256, 0, stream>>>(tbuf, outZ, Nn, Tt, NT * 128);
  };

  // ---- conv0 + bn + relu ----  (x=d_in[0])
  k_conv0<<<ew(NT * 128), 256, 0, stream>>>(P(0), P(2), P(3), P(4), P(5), P(6), P(7),
                                            cf, NT * 128);

  // ---- ev0: 4 resblocks (param bases 8,20,32,44) ----
  resblock(cf, cfB, 8);
  resblock(cfB, cf, 20);
  resblock(cf, cfB, 32);
  resblock(cfB, cf, 44);

  // ---- Z = transm(Cf) ----
  transm(cf, zb);

  // ---- Ol = dgcnn(Cf)  (neb params: w1=81,b1=82,bn1=83..86,w2=87,b2=88,bn2=89..92)
  k_topk<<<(unsigned)NT, 128, 0, stream>>>(cf, nbr, Nn);
  k_build_edge<<<ew(NT * 3 * 768), 256, 0, stream>>>(cf, nbr, G2b, Nn, NT * 3 * 768);
  gemm(G2b, P(81), P(82), nullptr, P(83), P(84), P(85), P(86),
       H1b, (int)(NT * 3), 128, 768, 4 | 1);
  k_pack_h2<<<ew(NT * 384), 256, 0, stream>>>(H1b, H2b, NT * 384);
  gemm(H2b, P(87), P(88), nullptr, P(89), P(90), P(91), P(92),
       ol, (int)NT, 128, 384, 4 | 1);

  // ---- ev1: 4 resblocks (param bases 93,105,117,129) ----
  resblock(ol, olB, 93);
  resblock(olB, ol, 105);
  resblock(ol, olB, 117);
  resblock(olB, ol, 129);

  // ---- out_l1 = cont1(concat(Z, Ol))  (cont1: w=141,b=142,bn=143..146) ----
  k_concat2<<<ew(NT * 256), 256, 0, stream>>>(zb, ol, catb, NT * 256);
  gemm(catb, P(141), P(142), nullptr, P(143), P(144), P(145), P(146),
       l1, (int)NT, 128, 256, 4 | 1);

  // ---- out_tm2 = transm(out_l1) ----
  transm(l1, tm2b);

  // ---- GCN branch (lin0 w=177,b=178; gcn w=159,b=160,bn=161..164) ----
  k_gemv<<<ew(NT), 256, 0, stream>>>(l1, P(177), P(178), wlb, (int)NT);
  k_gcn_prep<<<Bn, 256, 0, stream>>>(wlb, wab, rrb, Nn);
  k_gcn_tc<<<Bn * 128, 256, 0, stream>>>(l1, wab, rrb, tcb, Nn);
  k_gcn_apply<<<ew(NT * 128), 256, 0, stream>>>(l1, wab, rrb, tcb, goutb, Nn, NT * 128);
  gemm(goutb, P(159), P(160), nullptr, P(161), P(162), P(163), P(164),
       fgb, (int)NT, 128, 128, 4 | 1);
  k_add<<<ew(NT * 128), 256, 0, stream>>>(fgb, l1, fmb, NT * 128);
  resblock(fmb, ogb, 165);   // e1

  // ---- out_g1 = cont2(concat(tm2, Og))  (cont2: 147..152) ----
  k_concat2<<<ew(NT * 256), 256, 0, stream>>>(tm2b, ogb, catb, NT * 256);
  gemm(catb, P(147), P(148), nullptr, P(149), P(150), P(151), P(152),
       g1b, (int)NT, 128, 256, 4 | 1);

  // ---- out_tm3 = transm(out_g1) ----
  transm(g1b, tm3b);

  // ---- out_g2 = cont3(concat(tm3, Og))  (cont3: 153..158) ----
  k_concat2<<<ew(NT * 256), 256, 0, stream>>>(tm3b, ogb, catb, NT * 256);
  gemm(catb, P(153), P(154), nullptr, P(155), P(156), P(157), P(158),
       g2b, (int)NT, 128, 256, 4 | 1);

  // ---- w_g = lin1(out_g2) -> d_out[0:16000) ----
  float* outf = (float*)d_out;
  k_gemv<<<ew(NT), 256, 0, stream>>>(g2b, P(179), P(180), outf, (int)NT);

  // ---- downsample selection: x_ds @16000, y_ds @48000, w_ds @56000 ----
  k_select<<<Bn, 256, 0, stream>>>(outf, P(1), P(0),
                                   outf + 16000, outf + 48000, outf + 56000,
                                   Nn, 1000);
}
